// PerceiverCrossAttention_19559281066097
// MI455X (gfx1250) — matrix operand order repaired and checked by reading, not compiled, not run on hardware
//
#include <hip/hip_runtime.h>

// ---------------------------------------------------------------------------
// Perceiver cross-attention for MI455X (gfx1250, wave32, WMMA).
// LN->bf16, weight cast+transpose->bf16, Q GEMM, KV GEMM, V transpose,
// fused flash attention, output GEMM (fp32). All matmuls use
// v_wmma_f32_16x16x32_bf16. Tile staging uses the Tensor Data Mover
// (TENSOR_LOAD_TO_LDS with hardware LDS-row padding) when available,
// else GLOBAL_LOAD_ASYNC_TO_LDS_B128, else plain LDS stores.
// ---------------------------------------------------------------------------

#define DIM     3072
#define KV_DIM  2048
#define HEADS   16
#define DHEAD   128
#define INNER   2048
#define BATCH   2
#define SEQ_Q   2048
#define SEQ_KV  1024

typedef __attribute__((ext_vector_type(16))) __bf16 v16bf;
typedef __attribute__((ext_vector_type(8)))  __bf16 v8bf;
typedef __attribute__((ext_vector_type(8)))  float  v8f;
typedef __attribute__((ext_vector_type(4)))  int    v4i;
typedef __attribute__((ext_vector_type(8)))  int    v8i;
typedef __attribute__((ext_vector_type(4)))  unsigned int v4u;

// ---- feature detection ----------------------------------------------------
#if defined(__has_builtin)
#if __has_builtin(__builtin_amdgcn_tensor_load_to_lds) && \
    __has_builtin(__builtin_amdgcn_s_wait_tensorcnt)
#define HAVE_TDM 1
#endif
#if __has_builtin(__builtin_amdgcn_global_load_async_to_lds_b128)
#define HAVE_ASYNC_LDS 1
#endif
#if __has_builtin(__builtin_amdgcn_s_wait_asynccnt)
#define HAVE_WAIT_ASYNC 1
#endif
#endif
#ifndef HAVE_TDM
#define HAVE_TDM 0
#endif
#ifndef HAVE_ASYNC_LDS
#define HAVE_ASYNC_LDS 0
#endif
#ifndef HAVE_WAIT_ASYNC
#define HAVE_WAIT_ASYNC 0
#endif

#if HAVE_ASYNC_LDS
typedef __attribute__((address_space(1))) v4i* gv4i_p;  // global int4*
typedef __attribute__((address_space(3))) v4i* lv4i_p;  // LDS int4*
#endif
typedef __attribute__((address_space(3))) const void* lconst_p;

static __device__ __forceinline__ unsigned lds_offset(const void* p) {
  return (unsigned)(unsigned long long)(lconst_p)p;
}

static __device__ __forceinline__ void async_copy_b128(const __bf16* gsrc,
                                                       __bf16* lds_dst) {
#if HAVE_ASYNC_LDS
  __builtin_amdgcn_global_load_async_to_lds_b128((gv4i_p)(__bf16*)gsrc,
                                                 (lv4i_p)lds_dst, 0, 0);
#else
  *(v8bf*)lds_dst = *(const v8bf*)gsrc;
#endif
}

#if HAVE_TDM
// Issue a 2D TDM tile load: tile_rows rows of ex halves, global row stride
// (halves), hardware padding appends pad DWORDs (code) every interval-code
// DWORDs so LDS rows land at the padded stride the WMMA fragment loads use.
// D# layout per CDNA5 ISA 8.3/8.4 (group0: count/lds/global/type; group1:
// data_size, pad, tensor dims, tile dims, dim0 stride). This toolchain's
// builtin takes 6 operands: (g0, g1, g2, g3, g4, cpol).
static __device__ __forceinline__ void tdm_load_2d(const __bf16* gsrc,
                                                   unsigned lds_byte_off,
                                                   int ex, int rows,
                                                   int row_stride,
                                                   int pad_interval_code,
                                                   int pad_amount_code) {
  const unsigned long long ga = (unsigned long long)gsrc;
  v4u g0;
  g0.x = 1u;                                           // count=1, user mode
  g0.y = lds_byte_off;                                 // lds_addr (bytes)
  g0.z = (unsigned)(ga & 0xFFFFFFFFu);                 // global_addr[31:0]
  g0.w = (unsigned)((ga >> 32) & 0x1FFFFFFu) | (2u << 30);  // [56:32], type=2
  v8i g1;
  g1[0] = (1 << 16) |                                  // data_size = 2 bytes
          (1 << 20) |                                  // pad_enable
          (pad_interval_code << 22) | (pad_amount_code << 25);
  g1[1] = (ex & 0xFFFF) << 16;                         // tensor_dim0[15:0]
  g1[2] = ((unsigned)ex >> 16) | ((rows & 0xFFFF) << 16);  // dim0 hi, dim1 lo
  g1[3] = ((unsigned)rows >> 16) | ((ex & 0xFFFF) << 16);  // dim1 hi, tile_dim0
  g1[4] = rows & 0xFFFF;                               // tile_dim1 (tile_dim2=0)
  g1[5] = row_stride;                                  // tensor_dim0_stride lo
  g1[6] = 0;                                           // stride0 hi, stride1 lo
  g1[7] = 0;                                           // stride1 hi
  v4i g2; g2.x = g2.y = g2.z = g2.w = 0;
  v4i g3; g3.x = g3.y = g3.z = g3.w = 0;
  v8i g4;
#pragma unroll
  for (int i = 0; i < 8; ++i) g4[i] = 0;
  __builtin_amdgcn_tensor_load_to_lds(g0, g1, g2, g3, g4, 0);
}
#endif

template <int N>
static __device__ __forceinline__ void stage_wait() {
#if HAVE_TDM
  __builtin_amdgcn_s_wait_tensorcnt(N);
#elif HAVE_ASYNC_LDS
#if HAVE_WAIT_ASYNC
  __builtin_amdgcn_s_wait_asynccnt(N);
#else
  asm volatile("s_wait_asynccnt %0" ::"n"(N) : "memory");
#endif
#endif
}

// wait thresholds differ: TDM counts descriptors, async counts b128 loads
#if HAVE_TDM
#define GEMM_WNEXT 2
#define ATT_WNEXT 1
#else
#define GEMM_WNEXT 4
#define ATT_WNEXT 2
#endif

// ---- WMMA fragment helpers (ISA 7.12.2 layouts) ---------------------------
static __device__ __forceinline__ v16bf concat8(v8bf lo, v8bf hi) {
  return __builtin_shufflevector(lo, hi, 0,1,2,3,4,5,6,7,8,9,10,11,12,13,14,15);
}

// A fragment (16x32 bf16): lane L = row m = L&15; element e -> k =
// (e<8?0:16) + (L>>4)*8 + (e&7).
static __device__ __forceinline__ v16bf load_a_frag(const __bf16* row, int lhalf) {
  v8bf lo = *(const v8bf*)(row + lhalf * 8);
  v8bf hi = *(const v8bf*)(row + 16 + lhalf * 8);
  return concat8(lo, hi);
}

// B fragment (32x16 bf16) from an N-major LDS tile: lane L = column n = L&15;
// element e -> k = (e<8?0:16) + (L>>4)*8 + (e&7).
static __device__ __forceinline__ v16bf load_bt_frag(const __bf16* rowT, int lhalf) {
  const __bf16* p = rowT + lhalf * 8;
  v8bf lo = *(const v8bf*)(p);
  v8bf hi = *(const v8bf*)(p + 16);
  return concat8(lo, hi);
}

static __device__ __forceinline__ v8f wmma_bf16(v16bf a, v16bf b, v8f c) {
  return __builtin_amdgcn_wmma_f32_16x16x32_bf16(false, a, false, b, (short)0, c,
                                                 false, false);
}

// ---------------------------------------------------------------------------
// LayerNorm (fp32 in) -> bf16 out. One 256-thread block per row.
// ---------------------------------------------------------------------------
__global__ __launch_bounds__(256)
void layernorm_bf16_kernel(const float* __restrict__ x, const float* __restrict__ g,
                           const float* __restrict__ b, __bf16* __restrict__ out,
                           int n) {
  __shared__ float red[256];
  const int row = blockIdx.x;
  const int tid = threadIdx.x;
  const float* xr = x + (size_t)row * n;

  float s = 0.f;
  for (int i = tid; i < n; i += 256) s += xr[i];
  red[tid] = s; __syncthreads();
  for (int k = 128; k > 0; k >>= 1) { if (tid < k) red[tid] += red[tid + k]; __syncthreads(); }
  const float mu = red[0] / (float)n;
  __syncthreads();

  float v = 0.f;
  for (int i = tid; i < n; i += 256) { float d = xr[i] - mu; v += d * d; }
  red[tid] = v; __syncthreads();
  for (int k = 128; k > 0; k >>= 1) { if (tid < k) red[tid] += red[tid + k]; __syncthreads(); }
  const float rstd = rsqrtf(red[0] / (float)n + 1e-5f);

  __bf16* orow = out + (size_t)row * n;
  for (int i = tid; i < n; i += 256)
    orow[i] = (__bf16)((xr[i] - mu) * rstd * g[i] + b[i]);
}

// ---------------------------------------------------------------------------
// fp32 [K,N] -> bf16 [N,K] transpose+cast (weights), 32x32 LDS tiles.
// ---------------------------------------------------------------------------
__global__ __launch_bounds__(256)
void transpose_cast_kernel(const float* __restrict__ in, __bf16* __restrict__ out,
                           int K, int N) {
  __shared__ float t[32][33];
  const int k0 = blockIdx.y * 32, n0 = blockIdx.x * 32;
  const int tx = threadIdx.x & 31, ty = threadIdx.x >> 5;  // ty 0..7
#pragma unroll
  for (int i = 0; i < 4; ++i)
    t[ty + i * 8][tx] = in[(size_t)(k0 + ty + i * 8) * N + n0 + tx];
  __syncthreads();
#pragma unroll
  for (int i = 0; i < 4; ++i)
    out[(size_t)(n0 + ty + i * 8) * K + k0 + tx] = (__bf16)t[tx][ty + i * 8];
}

// ---------------------------------------------------------------------------
// V head-transpose: kv[b*SKV+s][INNER + h*128 + d] -> vt[((b*H+h)*128+d)*SKV+s]
// ---------------------------------------------------------------------------
__global__ __launch_bounds__(256)
void transpose_v_kernel(const __bf16* __restrict__ kv, __bf16* __restrict__ vt) {
  __shared__ __bf16 t[32][33];
  const int s0 = blockIdx.x * 32;
  const int d0 = (blockIdx.y & 3) * 32;
  const int h  = blockIdx.y >> 2;
  const int b  = blockIdx.z;
  const int tx = threadIdx.x & 31, ty = threadIdx.x >> 5;
#pragma unroll
  for (int i = 0; i < 4; ++i)
    t[ty + i * 8][tx] =
        kv[(size_t)(b * SEQ_KV + s0 + ty + i * 8) * (2 * INNER) + INNER + h * DHEAD + d0 + tx];
  __syncthreads();
#pragma unroll
  for (int i = 0; i < 4; ++i)
    vt[((size_t)(b * HEADS + h) * DHEAD + d0 + ty + i * 8) * SEQ_KV + s0 + tx] =
        t[tx][ty + i * 8];
}

// ---------------------------------------------------------------------------
// bf16 GEMM: C[M,N] = A[M,K] * Bt[N,K]^T (both row-major, K-contiguous).
// 256 threads / 8 waves; block tile 128x128, K-step 32; TDM (or async)
// double-buffered LDS staging.
// ---------------------------------------------------------------------------
#define TB_M 128
#define TB_N 128
#define TB_K 32
#define ALD  40   // halves per LDS row (32 + 8 pad) -> 80B stride (16B mult.)
#define BLD  40

template <bool OUT_F32>
__global__ __launch_bounds__(256)
void gemm_abt_kernel(const __bf16* __restrict__ A, const __bf16* __restrict__ Bt,
                     void* __restrict__ Cout, int M, int N, int K) {
  __shared__ alignas(16) __bf16 As[2][TB_M * ALD];
  __shared__ alignas(16) __bf16 Bs[2][TB_N * BLD];

  const int tid  = threadIdx.x;
  const int lane = tid & 31;
  const int wave = tid >> 5;
  const int lmod = lane & 15;
  const int lhalf = lane >> 4;
  const int m0 = blockIdx.y * TB_M;
  const int n0 = blockIdx.x * TB_N;
  const int wm = (wave >> 1) * 32;
  const int wn = (wave & 1) * 64;

  const int sr = tid >> 1;         // staging row 0..127 (async path)
  const int sc = (tid & 1) * 16;   // staging col 0 or 16 (async path)

  auto issue = [&](int k0, int buf) {
#if HAVE_TDM
    if (wave == 0) {
      // rows of 32 halves (16 DW) padded by 4 DW -> 40-half LDS stride
      tdm_load_2d(A + (size_t)m0 * K + k0, lds_offset(&As[buf][0]),
                  TB_K, TB_M, K, /*interval 16DW*/3, /*pad 4DW*/3);
      tdm_load_2d(Bt + (size_t)n0 * K + k0, lds_offset(&Bs[buf][0]),
                  TB_K, TB_M, K, 3, 3);
    }
#else
    const __bf16* gA = A + (size_t)(m0 + sr) * K + k0 + sc;
    async_copy_b128(gA,     &As[buf][sr * ALD + sc]);
    async_copy_b128(gA + 8, &As[buf][sr * ALD + sc + 8]);
    const __bf16* gB = Bt + (size_t)(n0 + sr) * K + k0 + sc;
    async_copy_b128(gB,     &Bs[buf][sr * BLD + sc]);
    async_copy_b128(gB + 8, &Bs[buf][sr * BLD + sc + 8]);
#endif
  };

  v8f acc[2][4];
#pragma unroll
  for (int i = 0; i < 2; ++i)
#pragma unroll
    for (int j = 0; j < 4; ++j)
#pragma unroll
      for (int r = 0; r < 8; ++r) acc[i][j][r] = 0.f;

  const int T = K / TB_K;
  issue(0, 0);
  for (int it = 0; it < T; ++it) {
    const int buf = it & 1;
    if (it + 1 < T) { issue((it + 1) * TB_K, buf ^ 1); stage_wait<GEMM_WNEXT>(); }
    else            { stage_wait<0>(); }
    __syncthreads();

    v16bf af[2], bf[4];
#pragma unroll
    for (int i = 0; i < 2; ++i)
      af[i] = load_a_frag(&As[buf][(wm + i * 16 + lmod) * ALD], lhalf);
#pragma unroll
    for (int j = 0; j < 4; ++j)
      bf[j] = load_bt_frag(&Bs[buf][(wn + j * 16 + lmod) * BLD], lhalf);
#pragma unroll
    for (int i = 0; i < 2; ++i)
#pragma unroll
      for (int j = 0; j < 4; ++j)
        acc[i][j] = wmma_bf16(af[i], bf[j], acc[i][j]);
    __syncthreads();
  }

#pragma unroll
  for (int i = 0; i < 2; ++i)
#pragma unroll
    for (int j = 0; j < 4; ++j) {
      const int n = n0 + wn + j * 16 + lmod;
#pragma unroll
      for (int r = 0; r < 8; ++r) {
        const int m = m0 + wm + i * 16 + r + 8 * lhalf;
        if (OUT_F32) ((float*)Cout)[(size_t)m * N + n] = acc[i][j][r];
        else        ((__bf16*)Cout)[(size_t)m * N + n] = (__bf16)acc[i][j][r];
      }
    }
}

// ---------------------------------------------------------------------------
// Fused flash attention. Block = (64 queries, head, batch); KV chunk 32.
// K tiles double-buffered; V (pre-transposed) pipelined single-buffer;
// online softmax in LDS. TDM (or async) staging.
// ---------------------------------------------------------------------------
#define AQB 64
#define AKB 32
#define QLD 136   // 272B stride
#define KLD 136
#define VLD 40
#define PLD 40
#define SLD 36

__global__ __launch_bounds__(256)
void attn_kernel(const __bf16* __restrict__ qw, const __bf16* __restrict__ kvw,
                 const __bf16* __restrict__ vt, __bf16* __restrict__ ow) {
  __shared__ alignas(16) __bf16 Qs[AQB * QLD];
  __shared__ alignas(16) __bf16 Ks[2][AKB * KLD];
  __shared__ alignas(16) __bf16 Vs[DHEAD * VLD];   // [d][kv]
  __shared__ alignas(16) __bf16 Ps[AQB * PLD];
  __shared__ alignas(16) float  Ss[AQB * SLD];
  __shared__ float mrow[AQB], lrow[AQB], crow[AQB];

  const int tid  = threadIdx.x;
  const int lane = tid & 31;
  const int wave = tid >> 5;
  const int lmod = lane & 15;
  const int lhalf = lane >> 4;
  const int q0 = blockIdx.x * AQB;
  const int h  = blockIdx.y;
  const int b  = blockIdx.z;
  const int qt = wave >> 1;
  const int j2 = wave & 1;
  const int dg = (wave & 1) * 64;

  auto issue_K = [&](int kv0, int kb) {
#if HAVE_TDM
    if (wave == 0)  // rows of 128 halves (64 DW) padded by 4 DW -> 136 stride
      tdm_load_2d(kvw + (size_t)(b * SEQ_KV + kv0) * (2 * INNER) + h * DHEAD,
                  lds_offset(&Ks[kb][0]), DHEAD, AKB, 2 * INNER, 5, 3);
#else
    const int r = tid >> 3, c = (tid & 7) * 16;
    const __bf16* g = kvw + (size_t)(b * SEQ_KV + kv0 + r) * (2 * INNER) + h * DHEAD + c;
    async_copy_b128(g,     &Ks[kb][r * KLD + c]);
    async_copy_b128(g + 8, &Ks[kb][r * KLD + c + 8]);
#endif
  };
  auto issue_V = [&](int kv0) {
#if HAVE_TDM
    if (wave == 0)  // rows of 32 halves (16 DW) padded by 4 DW -> 40 stride
      tdm_load_2d(vt + ((size_t)(b * HEADS + h) * DHEAD) * SEQ_KV + kv0,
                  lds_offset(&Vs[0]), AKB, DHEAD, SEQ_KV, 3, 3);
#else
    const int d = tid >> 1, c = (tid & 1) * 16;
    const __bf16* g = vt + ((size_t)(b * HEADS + h) * DHEAD + d) * SEQ_KV + kv0 + c;
    async_copy_b128(g,     &Vs[d * VLD + c]);
    async_copy_b128(g + 8, &Vs[d * VLD + c + 8]);
#endif
  };

  { // stage Q block [64 x 128] (waited together with first K/V)
#if HAVE_TDM
    if (wave == 0)
      tdm_load_2d(qw + (size_t)(b * SEQ_Q + q0) * INNER + h * DHEAD,
                  lds_offset(&Qs[0]), DHEAD, AQB, INNER, 5, 3);
#else
    const int r = tid >> 2, c = (tid & 3) * 32;
    const __bf16* g = qw + (size_t)(b * SEQ_Q + q0 + r) * INNER + h * DHEAD + c;
#pragma unroll
    for (int j = 0; j < 4; ++j)
      async_copy_b128(g + j * 8, &Qs[r * QLD + c + j * 8]);
#endif
  }
  issue_K(0, 0);
  issue_V(0);
  if (tid < AQB) { mrow[tid] = -__builtin_inff(); lrow[tid] = 0.f; }

  v8f o[4];
#pragma unroll
  for (int j = 0; j < 4; ++j)
#pragma unroll
    for (int r = 0; r < 8; ++r) o[j][r] = 0.f;

  const float scale2 = 0.08838834764831845f;  // 1/sqrt(128)
  const int T = SEQ_KV / AKB;

  for (int it = 0; it < T; ++it) {
    const int kb = it & 1;
    if (it + 1 < T) { issue_K((it + 1) * AKB, kb ^ 1); stage_wait<ATT_WNEXT>(); }
    else            { stage_wait<0>(); }
    __syncthreads();

    { // S tile (q 16-tile qt) x (kv 16-tile j2), K=128 in 4 WMMA steps
      v8f s;
#pragma unroll
      for (int r = 0; r < 8; ++r) s[r] = 0.f;
#pragma unroll
      for (int dk = 0; dk < 4; ++dk) {
        v16bf a  = load_a_frag(&Qs[(qt * 16 + lmod) * QLD + dk * 32], lhalf);
        v16bf bt = load_bt_frag(&Ks[kb][(j2 * 16 + lmod) * KLD + dk * 32], lhalf);
        s = wmma_bf16(a, bt, s);
      }
#pragma unroll
      for (int r = 0; r < 8; ++r)
        Ss[(qt * 16 + r + 8 * lhalf) * SLD + j2 * 16 + lmod] = s[r] * scale2;
    }
    __syncthreads();

    if (tid < AQB) { // online softmax, one thread per query row
      const float* srow = &Ss[tid * SLD];
      const float mold = mrow[tid];
      float mx = mold;
      float sv[AKB];
#pragma unroll
      for (int j = 0; j < AKB; ++j) { sv[j] = srow[j]; mx = fmaxf(mx, sv[j]); }
      const float corr = __expf(mold - mx);
      float sum = 0.f;
#pragma unroll
      for (int j = 0; j < AKB; ++j) {
        const float p = __expf(sv[j] - mx);
        sum += p;
        Ps[tid * PLD + j] = (__bf16)p;
      }
      lrow[tid] = lrow[tid] * corr + sum;
      mrow[tid] = mx;
      crow[tid] = corr;
    }
    __syncthreads();

    // rescale O, then O += P @ V
    float cf[8];
#pragma unroll
    for (int r = 0; r < 8; ++r) cf[r] = crow[qt * 16 + r + 8 * lhalf];
#pragma unroll
    for (int j = 0; j < 4; ++j)
#pragma unroll
      for (int r = 0; r < 8; ++r) o[j][r] *= cf[r];

    v16bf pa = load_a_frag(&Ps[(qt * 16 + lmod) * PLD], lhalf);
#pragma unroll
    for (int j = 0; j < 4; ++j) {
      v16bf vb = load_bt_frag(&Vs[(dg + j * 16 + lmod) * VLD], lhalf);
      o[j] = wmma_bf16(pa, vb, o[j]);
    }
    __syncthreads();                  // all waves done reading Vs/Ps
    if (it + 1 < T) issue_V((it + 1) * AKB);
  }

  float li[8];
#pragma unroll
  for (int r = 0; r < 8; ++r) {
    const float l = lrow[qt * 16 + r + 8 * lhalf];
    li[r] = (l > 0.f) ? (1.f / l) : 0.f;
  }
#pragma unroll
  for (int j = 0; j < 4; ++j) {
    const int d = h * DHEAD + dg + j * 16 + lmod;
#pragma unroll
    for (int r = 0; r < 8; ++r) {
      const int m = q0 + qt * 16 + r + 8 * lhalf;
      ow[(size_t)(b * SEQ_Q + m) * INNER + d] = (__bf16)(o[j][r] * li[r]);
    }
  }
}

// ---------------------------------------------------------------------------
// Host launcher
// ---------------------------------------------------------------------------
extern "C" void kernel_launch(void* const* d_in, const int* in_sizes, int n_in,
                              void* d_out, int out_size, void* d_ws, size_t ws_size,
                              hipStream_t stream) {
  (void)in_sizes; (void)n_in; (void)out_size; (void)ws_size;
  const float* image_embeds  = (const float*)d_in[0];
  const float* hidden_states = (const float*)d_in[1];
  const float* Wq   = (const float*)d_in[2];
  const float* Wkv  = (const float*)d_in[3];
  const float* Wout = (const float*)d_in[4];
  const float* ln1_g = (const float*)d_in[5];
  const float* ln1_b = (const float*)d_in[6];
  const float* ln2_g = (const float*)d_in[7];
  const float* ln2_b = (const float*)d_in[8];
  float* out = (float*)d_out;

  char* ws = (char*)d_ws;
  size_t off = 0;
  auto carve = [&](size_t elems) {
    __bf16* p = (__bf16*)(ws + off);
    off += (elems * sizeof(__bf16) + 255) & ~(size_t)255;
    return p;
  };
  __bf16* xq_b   = carve((size_t)BATCH * SEQ_Q * DIM);
  __bf16* xkv_b  = carve((size_t)BATCH * SEQ_KV * KV_DIM);
  __bf16* Wq_t   = carve((size_t)INNER * DIM);          // [2048,3072]
  __bf16* Wkv_t  = carve((size_t)2 * INNER * KV_DIM);   // [4096,2048]
  __bf16* Wout_t = carve((size_t)DIM * INNER);          // [3072,2048]
  __bf16* q_b    = carve((size_t)BATCH * SEQ_Q * INNER);
  __bf16* kv_b   = carve((size_t)BATCH * SEQ_KV * 2 * INNER);
  __bf16* vt_b   = carve((size_t)BATCH * HEADS * DHEAD * SEQ_KV);
  __bf16* at_b   = carve((size_t)BATCH * SEQ_Q * INNER);

  layernorm_bf16_kernel<<<BATCH * SEQ_KV, 256, 0, stream>>>(image_embeds, ln1_g, ln1_b, xkv_b, KV_DIM);
  layernorm_bf16_kernel<<<BATCH * SEQ_Q, 256, 0, stream>>>(hidden_states, ln2_g, ln2_b, xq_b, DIM);

  transpose_cast_kernel<<<dim3(INNER / 32, DIM / 32), 256, 0, stream>>>(Wq, Wq_t, DIM, INNER);
  transpose_cast_kernel<<<dim3(2 * INNER / 32, KV_DIM / 32), 256, 0, stream>>>(Wkv, Wkv_t, KV_DIM, 2 * INNER);
  transpose_cast_kernel<<<dim3(DIM / 32, INNER / 32), 256, 0, stream>>>(Wout, Wout_t, INNER, DIM);

  // q = xq @ Wq : [4096,3072] x [3072,2048]
  gemm_abt_kernel<false><<<dim3(INNER / TB_N, (BATCH * SEQ_Q) / TB_M), 256, 0, stream>>>(
      xq_b, Wq_t, (void*)q_b, BATCH * SEQ_Q, INNER, DIM);
  // kv = xkv @ Wkv : [2048,2048] x [2048,4096]
  gemm_abt_kernel<false><<<dim3((2 * INNER) / TB_N, (BATCH * SEQ_KV) / TB_M), 256, 0, stream>>>(
      xkv_b, Wkv_t, (void*)kv_b, BATCH * SEQ_KV, 2 * INNER, KV_DIM);

  transpose_v_kernel<<<dim3(SEQ_KV / 32, (DHEAD / 32) * HEADS, BATCH), 256, 0, stream>>>(kv_b, vt_b);

  attn_kernel<<<dim3(SEQ_Q / AQB, HEADS, BATCH), 256, 0, stream>>>(q_b, kv_b, vt_b, at_b);

  // out = attn @ Wout : [4096,2048] x [2048,3072] -> fp32
  gemm_abt_kernel<true><<<dim3(DIM / TB_N, (BATCH * SEQ_Q) / TB_M), 256, 0, stream>>>(
      at_b, Wout_t, (void*)out, BATCH * SEQ_Q, DIM, INNER);
}
